// Q2A_Function_721554506011
// MI455X (gfx1250) — matrix operand order repaired and checked by reading, not compiled, not run on hardware
//
#include <hip/hip_runtime.h>

typedef __attribute__((ext_vector_type(16))) _Float16 v16h;
typedef __attribute__((ext_vector_type(8)))  _Float16 v8h;
typedef __attribute__((ext_vector_type(2)))  _Float16 v2h;
typedef __attribute__((ext_vector_type(8)))  float    v8f;

#define NB 16
#define NT 1024
#define NP 48
#define NS 12
#define NA 24
#define ND 768
#define NDS 768
#define ND4 3072

// ---------------- WMMA GEMM: C[M,N] = act(A[M,K] @ W[K,N] + bias) -------------
// A, W, bias, C fp32 row-major; tiles converted to f16 in LDS; f32 accumulate.
// Block tile 128x128, K-step 32. 8 waves; each wave owns a 32x64 C tile
// (2x4 subtiles of 16x16) -> 8 WMMAs per K-step per wave.
// OOB rows/cols are handled by CLAMPING load indices (garbage lands only in
// padding rows/cols of C, which the store guard suppresses) -> no scalar
// edge path, staging is always vectorized global_load_b128.
#define BM 128
#define BN 128
#define BK 32
#define BKP 40   // LDS row pitch in halves (80B: 16B-aligned v8h loads)

__device__ __forceinline__ v16h frag_a(const _Float16* row, int half)
{
    // 16-bit A layout: elems 0..7 -> K = half*8 + 0..7 ; 8..15 -> 16 + half*8 + 0..7
    v8h lo = *(const v8h*)(row + half * 8);
    v8h hi = *(const v8h*)(row + 16 + half * 8);
    v16h a;
#pragma unroll
    for (int i = 0; i < 8; i++) { a[i] = lo[i]; a[i + 8] = hi[i]; }
    return a;
}

__device__ __forceinline__ v16h frag_b(const _Float16* row, int half)
{
    // 16-bit B layout: elems 0..7 -> K = half*8 + 0..7 ; 8..15 -> 16 + half*8 + 0..7  (row = BsT[n][.])
    v8h lo = *(const v8h*)(row + half * 8);
    v8h hi = *(const v8h*)(row + 16 + half * 8);
    v16h b;
#pragma unroll
    for (int i = 0; i < 8; i++) { b[i] = lo[i]; b[i + 8] = hi[i]; }
    return b;
}

__global__ __launch_bounds__(256)
void q2a_gemm_wmma(const float* __restrict__ A, const float* __restrict__ W,
                   const float* __restrict__ bias, float* __restrict__ C,
                   int M, int N, int K, int relu)
{
    __shared__ __align__(16) _Float16 As[BM][BKP];    // 10240 B
    __shared__ __align__(16) _Float16 BsT[BN][BKP];   // 10240 B  (B transposed [n][k])

    const int tid  = threadIdx.x;
    const int lane = tid & 31;
    const int wave = tid >> 5;
    const int bm = blockIdx.y * BM;
    const int bn = blockIdx.x * BN;
    const int wm = (wave & 3) * 32;    // 32-row band  (2 subtiles)
    const int wn = (wave >> 2) * 64;   // 64-col band  (4 subtiles)
    const int ml   = lane & 15;
    const int half = lane >> 4;

    v8f acc[2][4];
#pragma unroll
    for (int i = 0; i < 2; i++)
#pragma unroll
        for (int j = 0; j < 4; j++) acc[i][j] = (v8f){};

    // clamped staging coordinates (uniform per thread across the K loop)
    int a_r0, a_c0, a_r1, a_c1;
    {
        int slot0 = tid, slot1 = tid + 256;
        int r0 = bm + (slot0 >> 2); a_r0 = (r0 < M) ? r0 : (M - 1);
        int r1 = bm + (slot1 >> 2); a_r1 = (r1 < M) ? r1 : (M - 1);
        a_c0 = (slot0 & 3) * 8;
        a_c1 = (slot1 & 3) * 8;
    }
    const int b_k = (tid >> 4) * 2;          // k-row pair within tile
    int b_n;
    {
        int n = bn + (tid & 15) * 8;
        b_n = (n <= N - 8) ? n : (N - 8);    // clamp (N >= 8 always here)
    }
    const int b_ln = (tid & 15) * 8;         // local n for LDS store

    for (int k0 = 0; k0 < K; k0 += BK) {
        // ---------------- stage A tile (BM x BK), 8 floats/thread x2 ----------
        {
            const float* src0 = &A[(size_t)a_r0 * K + k0 + a_c0];
            const float* src1 = &A[(size_t)a_r1 * K + k0 + a_c1];
            float4 f0 = *(const float4*)src0;
            float4 f1 = *(const float4*)(src0 + 4);
            float4 g0 = *(const float4*)src1;
            float4 g1 = *(const float4*)(src1 + 4);
            v8h h0, h1;
            h0[0] = (_Float16)f0.x; h0[1] = (_Float16)f0.y;
            h0[2] = (_Float16)f0.z; h0[3] = (_Float16)f0.w;
            h0[4] = (_Float16)f1.x; h0[5] = (_Float16)f1.y;
            h0[6] = (_Float16)f1.z; h0[7] = (_Float16)f1.w;
            h1[0] = (_Float16)g0.x; h1[1] = (_Float16)g0.y;
            h1[2] = (_Float16)g0.z; h1[3] = (_Float16)g0.w;
            h1[4] = (_Float16)g1.x; h1[5] = (_Float16)g1.y;
            h1[6] = (_Float16)g1.z; h1[7] = (_Float16)g1.w;
            *(v8h*)&As[tid >> 2][a_c0] = h0;
            *(v8h*)&As[(tid + 256) >> 2][a_c1] = h1;
        }
        // ---------------- stage W tile (BK x BN) transposed -------------------
        {
            const float* s0 = &W[(size_t)(k0 + b_k) * N + b_n];
            const float* s1 = s0 + N;
            float4 a0 = *(const float4*)s0, a1 = *(const float4*)(s0 + 4);
            float4 b0 = *(const float4*)s1, b1 = *(const float4*)(s1 + 4);
            float r0[8] = {a0.x, a0.y, a0.z, a0.w, a1.x, a1.y, a1.z, a1.w};
            float r1[8] = {b0.x, b0.y, b0.z, b0.w, b1.x, b1.y, b1.z, b1.w};
#pragma unroll
            for (int j = 0; j < 8; j++) {
                v2h p;
                p[0] = (_Float16)r0[j];
                p[1] = (_Float16)r1[j];
                *(v2h*)&BsT[b_ln + j][b_k] = p;
            }
        }
        // speculative prefetch of next A tile (global_prefetch_b8)
        if (k0 + BK < K) {
            __builtin_prefetch(&A[(size_t)a_r0 * K + k0 + BK + a_c0], 0, 1);
        }
        __syncthreads();

        // ---------------- compute: preload all fragments, then 8 WMMAs --------
        v16h fa0 = frag_a(&As[wm + ml][0], half);
        v16h fa1 = frag_a(&As[wm + 16 + ml][0], half);
        v16h fb0 = frag_b(&BsT[wn + ml][0], half);
        v16h fb1 = frag_b(&BsT[wn + 16 + ml][0], half);
        v16h fb2 = frag_b(&BsT[wn + 32 + ml][0], half);
        v16h fb3 = frag_b(&BsT[wn + 48 + ml][0], half);
        acc[0][0] = __builtin_amdgcn_wmma_f32_16x16x32_f16(false, fa0, false, fb0, (short)0, acc[0][0], false, false);
        acc[1][0] = __builtin_amdgcn_wmma_f32_16x16x32_f16(false, fa1, false, fb0, (short)0, acc[1][0], false, false);
        acc[0][1] = __builtin_amdgcn_wmma_f32_16x16x32_f16(false, fa0, false, fb1, (short)0, acc[0][1], false, false);
        acc[1][1] = __builtin_amdgcn_wmma_f32_16x16x32_f16(false, fa1, false, fb1, (short)0, acc[1][1], false, false);
        acc[0][2] = __builtin_amdgcn_wmma_f32_16x16x32_f16(false, fa0, false, fb2, (short)0, acc[0][2], false, false);
        acc[1][2] = __builtin_amdgcn_wmma_f32_16x16x32_f16(false, fa1, false, fb2, (short)0, acc[1][2], false, false);
        acc[0][3] = __builtin_amdgcn_wmma_f32_16x16x32_f16(false, fa0, false, fb3, (short)0, acc[0][3], false, false);
        acc[1][3] = __builtin_amdgcn_wmma_f32_16x16x32_f16(false, fa1, false, fb3, (short)0, acc[1][3], false, false);
        __syncthreads();
    }

    // C/D layout: VGPR r -> row = r + half*8 within subtile, col = lane&15
#pragma unroll
    for (int i = 0; i < 2; i++) {
#pragma unroll
        for (int r = 0; r < 8; r++) {
            int row = bm + wm + i * 16 + r + half * 8;
            if (row >= M) continue;
#pragma unroll
            for (int j = 0; j < 4; j++) {
                int col = bn + wn + j * 16 + ml;
                if (col < N) {
                    float v = acc[i][j][r] + bias[col];
                    if (relu) v = fmaxf(v, 0.0f);
                    C[(size_t)row * N + col] = v;
                }
            }
        }
    }
}

// ---------------- elementwise / reduction kernels -----------------------------

__global__ void q2a_softmax_rows(const float* __restrict__ in, float* __restrict__ out, int L)
{
    int b = blockIdx.x;
    int t = threadIdx.x;
    __shared__ float sh[64];
    float v = (t < L) ? in[b * L + t] : -1e30f;
    sh[t] = v; __syncthreads();
    for (int s = 32; s > 0; s >>= 1) { if (t < s) sh[t] = fmaxf(sh[t], sh[t + s]); __syncthreads(); }
    float mx = sh[0]; __syncthreads();
    float e = (t < L) ? __expf(v - mx) : 0.0f;
    sh[t] = e; __syncthreads();
    for (int s = 32; s > 0; s >>= 1) { if (t < s) sh[t] += sh[t + s]; __syncthreads(); }
    float sum = sh[0];
    if (t < L) out[b * L + t] = e / sum;
}

// in-place inclusive prefix sum along T; one thread per (b,d); coalesced in d
__global__ void q2a_cumsum_T(float* __restrict__ v, int Bn, int T, int Dn)
{
    int idx = blockIdx.x * blockDim.x + threadIdx.x;
    if (idx >= Bn * Dn) return;
    int b = idx / Dn, d = idx % Dn;
    size_t base = (size_t)b * T * Dn + d;
    float s = 0.0f;
    for (int t = 0; t < T; t++) {
        s += v[base + (size_t)t * Dn];
        v[base + (size_t)t * Dn] = s;
    }
}

// seg[b,p,d]: segment mean via inclusive prefix sums, or first element if s>=e
__global__ void q2a_seg(const float* __restrict__ incl, const int* __restrict__ starts,
                        const int* __restrict__ ends, float* __restrict__ seg,
                        int Bn, int Pn, int T, int Dn)
{
    size_t idx = (size_t)blockIdx.x * blockDim.x + threadIdx.x;
    if (idx >= (size_t)Bn * Pn * Dn) return;
    int d = (int)(idx % Dn);
    int bp = (int)(idx / Dn);
    int p = bp % Pn, b = bp / Pn;
    int s = starts[b * Pn + p], e = ends[b * Pn + p];
    size_t base = (size_t)b * T * Dn + d;
    float cs = (s > 0) ? incl[base + (size_t)(s - 1) * Dn] : 0.0f;
    float ce = (e > 0) ? incl[base + (size_t)(e - 1) * Dn] : 0.0f;
    int dn = e - s; if (dn < 1) dn = 1;
    float mean = (ce - cs) / (float)dn;
    float first = incl[base + (size_t)s * Dn] - cs;
    seg[idx] = (s >= e) ? first : mean;
}

// out[b,d] = sum_p score[b,p] * X[b,p,d]
__global__ void q2a_wsum_p(const float* __restrict__ score, const float* __restrict__ X,
                           float* __restrict__ out, int Bn, int Pn, int Dn)
{
    int idx = blockIdx.x * blockDim.x + threadIdx.x;
    if (idx >= Bn * Dn) return;
    int d = idx % Dn, b = idx / Dn;
    float acc = 0.0f;
    for (int p = 0; p < Pn; p++)
        acc += score[b * Pn + p] * X[((size_t)b * Pn + p) * Dn + d];
    out[idx] = acc;
}

// inputs[b,s,a,:] = [vseg(b) | tseg(b) | q(b)+at(b,s,a) | ab(b,s,a)]
__global__ void q2a_concat(const float* __restrict__ vseg, const float* __restrict__ tseg,
                           const float* __restrict__ q, const float* __restrict__ at,
                           const float* __restrict__ ab, float* __restrict__ out,
                           int Bn, int Sn, int An, int Dn)
{
    size_t total = (size_t)Bn * Sn * An * 4 * Dn;
    size_t idx = (size_t)blockIdx.x * blockDim.x + threadIdx.x;
    if (idx >= total) return;
    int c = (int)(idx % (4 * Dn));
    size_t row = idx / (4 * Dn);
    size_t bs = row / An;
    int b = (int)(bs / Sn);
    int d = c % Dn, chunk = c / Dn;
    float v;
    if (chunk == 0)      v = vseg[b * Dn + d];
    else if (chunk == 1) v = tseg[b * Dn + d];
    else if (chunk == 2) v = q[b * Dn + d] + at[row * Dn + d];
    else                 v = ab[row * Dn + d];
    out[idx] = v;
}

__global__ void q2a_bcast_state(const float* __restrict__ s0, float* __restrict__ h,
                                int Bn, int Dn)
{
    int idx = blockIdx.x * blockDim.x + threadIdx.x;
    if (idx >= Bn * Dn) return;
    h[idx] = s0[idx % Dn];
}

// GRU gating for time step s: hnew[b,a,:] from gi_all[b,s,a,:], gh[b,:], h[b,:]
__global__ void q2a_gru_gate(const float* __restrict__ gi_all, const float* __restrict__ gh,
                             const float* __restrict__ h, float* __restrict__ hnew,
                             int s, int Bn, int Sn, int An, int DSn)
{
    int idx = blockIdx.x * blockDim.x + threadIdx.x;
    if (idx >= Bn * An * DSn) return;
    int j = idx % DSn;
    int ba = idx / DSn;
    int a = ba % An, b = ba / An;
    size_t gir = ((size_t)(b * Sn + s) * An + a) * (3 * DSn);
    float ir = gi_all[gir + j];
    float iz = gi_all[gir + DSn + j];
    float in_ = gi_all[gir + 2 * DSn + j];
    size_t ghr = (size_t)b * (3 * DSn);
    float hr = gh[ghr + j];
    float hz = gh[ghr + DSn + j];
    float hn = gh[ghr + 2 * DSn + j];
    float r = 1.0f / (1.0f + __expf(-(ir + hr)));
    float z = 1.0f / (1.0f + __expf(-(iz + hz)));
    float n = tanhf(in_ + r * hn);
    hnew[idx] = (1.0f - z) * n + z * h[b * DSn + j];
}

// logits[row] = X[row,:] . w + bias ; one wave32 per row
__global__ void q2a_rowdot(const float* __restrict__ X, const float* __restrict__ w,
                           const float* __restrict__ bias, float* __restrict__ out,
                           int Rn, int Kn)
{
    int wave = threadIdx.x >> 5, lane = threadIdx.x & 31;
    int row = blockIdx.x * (blockDim.x >> 5) + wave;
    if (row >= Rn) return;
    float acc = 0.0f;
    for (int k = lane; k < Kn; k += 32) acc += X[(size_t)row * Kn + k] * w[k];
    for (int off = 16; off > 0; off >>= 1) acc += __shfl_xor(acc, off, 32);
    if (lane == 0) out[row] = acc + bias[0];
}

// per-b: NLL loss for step s + copy next hidden state from selected answer
__global__ void q2a_loss_next(const float* __restrict__ logits, const int* __restrict__ label,
                              const float* __restrict__ x_all, float* __restrict__ loss,
                              float* __restrict__ h, int s, int Bn, int Sn, int An, int DSn)
{
    int b = blockIdx.x;
    int lab = label[b * Sn + s];
    if (threadIdx.x == 0) {
        float mx = -1e30f;
        for (int a = 0; a < An; a++) mx = fmaxf(mx, logits[b * An + a]);
        float sum = 0.0f;
        for (int a = 0; a < An; a++) sum += __expf(logits[b * An + a] - mx);
        float lse = mx + __logf(sum);
        loss[s * Bn + b] = -(logits[b * An + lab] - lse);
    }
    size_t src = ((size_t)(b * Sn + s) * An + lab) * DSn;
    for (int j = threadIdx.x; j < DSn; j += blockDim.x)
        h[b * DSn + j] = x_all[src + j];
}

__global__ void q2a_mean(const float* __restrict__ loss, float* __restrict__ out, int n)
{
    __shared__ float sh[256];
    float acc = 0.0f;
    for (int i = threadIdx.x; i < n; i += 256) acc += loss[i];
    sh[threadIdx.x] = acc; __syncthreads();
    for (int s = 128; s > 0; s >>= 1) {
        if (threadIdx.x < s) sh[threadIdx.x] += sh[threadIdx.x + s];
        __syncthreads();
    }
    if (threadIdx.x == 0) out[0] = sh[0] / (float)n;
}

// ---------------- host orchestration -----------------------------------------

extern "C" void kernel_launch(void* const* d_in, const int* in_sizes, int n_in,
                              void* d_out, int out_size, void* d_ws, size_t ws_size,
                              hipStream_t stream)
{
    (void)in_sizes; (void)n_in; (void)out_size; (void)ws_size;
    const float* video    = (const float*)d_in[0];
    const float* para     = (const float*)d_in[1];
    const float* question = (const float*)d_in[2];
    const float* a_texts  = (const float*)d_in[3];
    const float* a_buttons= (const float*)d_in[4];
    const float* pscore   = (const float*)d_in[5];
    const int*   starts   = (const int*)d_in[6];
    const int*   ends     = (const int*)d_in[7];
    const int*   label    = (const int*)d_in[8];
    const float* wv1=(const float*)d_in[9],  *bv1=(const float*)d_in[10];
    const float* wv2=(const float*)d_in[11], *bv2=(const float*)d_in[12];
    const float* wt1=(const float*)d_in[13], *bt1=(const float*)d_in[14];
    const float* wt2=(const float*)d_in[15], *bt2=(const float*)d_in[16];
    const float* wp1=(const float*)d_in[17], *bp1=(const float*)d_in[18];
    const float* wp2=(const float*)d_in[19], *bp2=(const float*)d_in[20];
    const float* w_ih=(const float*)d_in[21],*b_ih=(const float*)d_in[22];
    const float* w_hh=(const float*)d_in[23],*b_hh=(const float*)d_in[24];
    const float* wq1=(const float*)d_in[25], *bq1=(const float*)d_in[26];
    const float* wq2=(const float*)d_in[27], *bq2=(const float*)d_in[28];
    const float* state0=(const float*)d_in[29];

    float* ws = (float*)d_ws;
    size_t o = 0;
    const size_t BT_D = (size_t)NB * NT * ND;
    const size_t BP_D = (size_t)NB * NP * ND;
    const size_t B_D  = (size_t)NB * ND;
    const size_t R    = (size_t)NB * NS * NA;   // 4608 answer rows

    float* vid_h1  = ws + o; o += BT_D;
    float* vid_v   = ws + o; o += BT_D;
    float* para_h1 = ws + o; o += BP_D;
    float* para_t  = ws + o; o += BP_D;
    float* seg     = ws + o; o += BP_D;
    float* at_buf  = ws + o; o += R * ND;
    float* ab_buf  = ws + o; o += R * ND;
    float* inputs  = ws + o; o += R * ND4;
    float* xh1     = ws + o; o += R * ND4;
    float* xall    = ws + o; o += R * NDS;
    float* q_h1    = ws + o; o += B_D;
    float* qv      = ws + o; o += B_D;
    float* score   = ws + o; o += (size_t)NB * NP;
    float* tseg    = ws + o; o += B_D;
    float* vseg    = ws + o; o += B_D;
    float* hbuf    = ws + o; o += B_D;
    float* gh      = ws + o; o += (size_t)NB * 3 * NDS;
    float* hnew    = ws + o; o += (size_t)NB * NA * NDS;
    float* hq1     = ws + o; o += (size_t)NB * NA * NDS;
    float* logits  = ws + o; o += (size_t)NB * NA;
    float* lossb   = ws + o; o += (size_t)NS * NB;
    // reused regions (stream ordering guarantees prior consumers are done)
    float* at_h1 = vid_h1;               // vid_h1 dead after vid_v GEMM
    float* ab_h1 = vid_h1 + R * ND;
    float* giall = inputs;               // inputs dead after xh1 GEMM

    auto gemm = [&](const float* Am, const float* Wm, const float* bm_,
                    float* Cm, int M, int N, int K, int relu) {
        dim3 g((N + BN - 1) / BN, (M + BM - 1) / BM);
        q2a_gemm_wmma<<<g, 256, 0, stream>>>(Am, Wm, bm_, Cm, M, N, K, relu);
    };

    // paragraph score softmax
    q2a_softmax_rows<<<NB, 64, 0, stream>>>(pscore, score, NP);
    // text branch: para MLP + weighted sum
    gemm(para, wt1, bt1, para_h1, NB * NP, ND, ND, 1);
    gemm(para_h1, wt2, bt2, para_t, NB * NP, ND, ND, 0);
    q2a_wsum_p<<<(NB * ND + 255) / 256, 256, 0, stream>>>(score, para_t, tseg, NB, NP, ND);
    // video branch: video MLP, prefix sums, segment means, weighted sum
    gemm(video, wv1, bv1, vid_h1, NB * NT, ND, ND, 1);
    gemm(vid_h1, wv2, bv2, vid_v, NB * NT, ND, ND, 0);
    q2a_cumsum_T<<<(NB * ND + 255) / 256, 256, 0, stream>>>(vid_v, NB, NT, ND);
    q2a_seg<<<(int)((BP_D + 255) / 256), 256, 0, stream>>>(vid_v, starts, ends, seg, NB, NP, NT, ND);
    q2a_wsum_p<<<(NB * ND + 255) / 256, 256, 0, stream>>>(score, seg, vseg, NB, NP, ND);
    // question MLP
    gemm(question, wt1, bt1, q_h1, NB, ND, ND, 1);
    gemm(q_h1, wt2, bt2, qv, NB, ND, ND, 0);
    // answer text / button MLPs (4608 rows each)
    gemm(a_texts, wt1, bt1, at_h1, (int)R, ND, ND, 1);
    gemm(at_h1, wt2, bt2, at_buf, (int)R, ND, ND, 0);
    gemm(a_buttons, wv1, bv1, ab_h1, (int)R, ND, ND, 1);
    gemm(ab_h1, wv2, bv2, ab_buf, (int)R, ND, ND, 0);
    // fusion MLP: concat -> 3072x3072 -> 3072x768
    q2a_concat<<<(int)((R * ND4 + 255) / 256), 256, 0, stream>>>(
        vseg, tseg, qv, at_buf, ab_buf, inputs, NB, NS, NA, ND);
    gemm(inputs, wp1, bp1, xh1, (int)R, ND4, ND4, 1);
    gemm(xh1, wp2, bp2, xall, (int)R, NDS, ND4, 0);
    // GRU input projection for all timesteps at once
    gemm(xall, w_ih, b_ih, giall, (int)R, 3 * NDS, NDS, 0);
    // GRU scan
    q2a_bcast_state<<<(NB * NDS + 255) / 256, 256, 0, stream>>>(state0, hbuf, NB, NDS);
    for (int s = 0; s < NS; s++) {
        gemm(hbuf, w_hh, b_hh, gh, NB, 3 * NDS, NDS, 0);
        q2a_gru_gate<<<(NB * NA * NDS + 255) / 256, 256, 0, stream>>>(
            giall, gh, hbuf, hnew, s, NB, NS, NA, NDS);
        gemm(hnew, wq1, bq1, hq1, NB * NA, NDS, NDS, 1);
        q2a_rowdot<<<(NB * NA + 3) / 4, 128, 0, stream>>>(hq1, wq2, bq2, logits, NB * NA, NDS);
        q2a_loss_next<<<NB, 256, 0, stream>>>(logits, label, xall, lossb, hbuf,
                                              s, NB, NS, NA, NDS);
    }
    q2a_mean<<<1, 256, 0, stream>>>(lossb, (float*)d_out, NS * NB);
}
